// TTLinear_61211873903091
// MI455X (gfx1250) — matrix operand order repaired and checked by reading, not compiled, not run on hardware
//
#include <hip/hip_runtime.h>
#include <cstdint>

// ---------------------------------------------------------------------------
// TT-linear on MI455X (gfx1250, wave32, WMMA).
//
// (1) Expand the tiny TT cores into dense bf16 W[4096][4096] in workspace
//     (~1 GMAC; 32 MB -> L2-resident, 192 MB L2).
// (2) Dense GEMM Y[8192x4096] = X(f32->bf16) * W^T with
//     v_wmma_f32_16x16x32_bf16, f32 accumulate, bias epilogue.
//     Block tile 128x256, 8 wave32 each owning 64x64 (16 WMMA tiles).
//     W tiles staged with GLOBAL_LOAD_ASYNC_TO_LDS (ASYNCcnt), X staged
//     via VGPRs with packed bf16 conversion. HBM floor ~11 us; compute
//     bound at 274 GFLOP bf16.
// ---------------------------------------------------------------------------

typedef __attribute__((ext_vector_type(16))) __bf16 v16bf;
typedef __attribute__((ext_vector_type(8)))  __bf16 v8bf;
typedef __attribute__((ext_vector_type(8)))  float  v8f;
typedef __attribute__((ext_vector_type(4)))  int    v4i;

#define MDIM 8192
#define NDIM 4096
#define KDIM 4096
#define BM 128
#define BN 256
#define BK 32
#define LDK 40   // padded LDS row stride (bf16): 80 B -> conflict-free frag reads
#define KSTEPS (KDIM / BK)

#if defined(__gfx1250__) && __has_builtin(__builtin_amdgcn_global_load_async_to_lds_b128)
#define USE_ASYNC_LDS 1
#else
#define USE_ASYNC_LDS 0
#endif

__device__ __forceinline__ void async_cp_b128(const void* g, void* l)
{
#if USE_ASYNC_LDS
    // param0: AS1 (global) v4i*, param1: AS3 (LDS) v4i*, then imm offset, imm cpol
    __builtin_amdgcn_global_load_async_to_lds_b128(
        (__attribute__((address_space(1))) v4i*)(uintptr_t)g,
        (__attribute__((address_space(3))) v4i*)(unsigned)(uintptr_t)l,
        0, 0);
#else
    (void)g; (void)l;
#endif
}

__device__ __forceinline__ void wait_async0()
{
#if USE_ASYNC_LDS
#if __has_builtin(__builtin_amdgcn_s_wait_asynccnt)
    __builtin_amdgcn_s_wait_asynccnt(0);
#else
    asm volatile("s_wait_asynccnt 0x0" ::: "memory");
#endif
#endif
}

// ---------------------------------------------------------------------------
// Kernel 1: expand TT cores -> dense bf16 W[o][i].
// ---------------------------------------------------------------------------
__global__ __launch_bounds__(256) void tt_expand_w(
    const float* __restrict__ c0, const float* __restrict__ c1,
    const float* __restrict__ c2, __bf16* __restrict__ W)
{
    unsigned tid = blockIdx.x * 256u + threadIdx.x;   // 0 .. 16M-1
    unsigned o = tid >> 12;
    unsigned i = tid & 4095u;
    unsigned o0 = o >> 8, o1 = (o >> 4) & 15u, o2 = o & 15u;
    unsigned i0 = i >> 8, i1 = (i >> 4) & 15u, i2 = i & 15u;

    float t[8];
#pragma unroll
    for (int r2 = 0; r2 < 8; ++r2) t[r2] = 0.f;

    const float* c0p = c0 + (o0 * 16u + i0) * 8u;
#pragma unroll
    for (int r1 = 0; r1 < 8; ++r1) {
        float a0 = c0p[r1];
        const float* c1p = c1 + (((unsigned)r1 * 16u + o1) * 16u + i1) * 8u;
#pragma unroll
        for (int r2 = 0; r2 < 8; ++r2) t[r2] = fmaf(a0, c1p[r2], t[r2]);
    }
    float acc = 0.f;
#pragma unroll
    for (int r2 = 0; r2 < 8; ++r2)
        acc = fmaf(t[r2], c2[((unsigned)r2 * 16u + o2) * 16u + i2], acc);

    W[(size_t)o * KDIM + i] = (__bf16)acc;
}

// ---------------------------------------------------------------------------
// Kernel 2: dense GEMM, bf16 WMMA, f32 accumulate, bias epilogue.
// ---------------------------------------------------------------------------
__device__ __forceinline__ v8bf cvt8(float4 a, float4 b)
{
    v8bf r;
    r[0] = (__bf16)a.x; r[1] = (__bf16)a.y; r[2] = (__bf16)a.z; r[3] = (__bf16)a.w;
    r[4] = (__bf16)b.x; r[5] = (__bf16)b.y; r[6] = (__bf16)b.z; r[7] = (__bf16)b.w;
    return r;
}

__device__ __forceinline__ v16bf cat16(v8bf lo, v8bf hi)
{
    return __builtin_shufflevector(lo, hi, 0,1,2,3,4,5,6,7,8,9,10,11,12,13,14,15);
}

__global__ __launch_bounds__(256) void tt_gemm_bf16(
    const float* __restrict__ X, const __bf16* __restrict__ W,
    const float* __restrict__ bias, float* __restrict__ Y)
{
    __shared__ __align__(16) __bf16 Xs[2][BM][LDK];
    __shared__ __align__(16) __bf16 Ws[2][BN][LDK];

    const int t    = threadIdx.x;
    const int lane = t & 31;
    const int wave = t >> 5;
    const int wm   = (wave >> 2) * 64;   // wave row base (0/64)
    const int wn   = (wave & 3) * 64;    // wave col base (0..192)
    const int m0   = lane & 15;          // row (A) / col (B,D) within 16x16 tile
    const int ks   = lane >> 4;          // K-half selector (ISA fragment layout)
    const int bM   = blockIdx.y * BM;
    const int bN   = blockIdx.x * BN;

    // X staging role: (row 0..127, 16-elem K chunk 0/16)
    const int r  = t >> 1;
    const int ch = (t & 1) * 16;
    const float* xg = X + (size_t)(bM + r) * KDIM + ch;

    // W staging role: one row (bN + t), 32 bf16 = 4x16B per K-step
    const __bf16* wgrow = W + (size_t)(bN + t) * KDIM;

    v8f acc[4][4] = {};

    // ---- prologue: stage K-tile 0 into buffer 0 ----
    {
#if USE_ASYNC_LDS
#pragma unroll
        for (int c = 0; c < 4; ++c)
            async_cp_b128(wgrow + c * 8, &Ws[0][t][c * 8]);
#else
        const v8bf* wp = (const v8bf*)wgrow;
#pragma unroll
        for (int c = 0; c < 4; ++c)
            *(v8bf*)&Ws[0][t][c * 8] = wp[c];
#endif
        const float4* xp = (const float4*)xg;
        float4 x0 = xp[0], x1 = xp[1], x2 = xp[2], x3 = xp[3];
        *(v8bf*)&Xs[0][r][ch]     = cvt8(x0, x1);
        *(v8bf*)&Xs[0][r][ch + 8] = cvt8(x2, x3);
    }
    wait_async0();
    __syncthreads();

    // ---- main K loop, double buffered, one barrier per step ----
    for (int kt = 0; kt < KSTEPS; ++kt) {
        const int cur = kt & 1;
        const int nxt = cur ^ 1;
        const bool pref = (kt + 1) < KSTEPS;

        float4 x0, x1, x2, x3;
#if !USE_ASYNC_LDS
        v8bf wreg[4];
#endif
        if (pref) {
#if USE_ASYNC_LDS
            // background copy of next W tile straight into LDS (ASYNCcnt)
            const __bf16* wp = wgrow + (size_t)(kt + 1) * BK;
#pragma unroll
            for (int c = 0; c < 4; ++c)
                async_cp_b128(wp + c * 8, &Ws[nxt][t][c * 8]);
#else
            const v8bf* wp = (const v8bf*)(wgrow + (size_t)(kt + 1) * BK);
#pragma unroll
            for (int c = 0; c < 4; ++c) wreg[c] = wp[c];
#endif
            const float4* xp = (const float4*)(xg + (size_t)(kt + 1) * BK);
            x0 = xp[0]; x1 = xp[1]; x2 = xp[2]; x3 = xp[3];
        }

        // A fragments: lane = row m0; K = ks*8..+7 and 16+ks*8..+7
        v16bf af[4];
#pragma unroll
        for (int i = 0; i < 4; ++i) {
            const __bf16* p = &Xs[cur][wm + i * 16 + m0][ks * 8];
            af[i] = cat16(*(const v8bf*)p, *(const v8bf*)(p + 16));
        }
        // B fragments: lane = col m0; K = ks*8..+7 and 16+ks*8..+7
        v16bf bfrag[4];
#pragma unroll
        for (int j = 0; j < 4; ++j) {
            const __bf16* p = &Ws[cur][wn + j * 16 + m0][ks * 8];
            bfrag[j] = cat16(*(const v8bf*)p, *(const v8bf*)(p + 16));
        }

#pragma unroll
        for (int i = 0; i < 4; ++i)
#pragma unroll
            for (int j = 0; j < 4; ++j)
                acc[i][j] = __builtin_amdgcn_wmma_f32_16x16x32_bf16(
                    false, af[i], false, bfrag[j], (short)0, acc[i][j], false, false);

        if (pref) {
            *(v8bf*)&Xs[nxt][r][ch]     = cvt8(x0, x1);
            *(v8bf*)&Xs[nxt][r][ch + 8] = cvt8(x2, x3);
#if !USE_ASYNC_LDS
#pragma unroll
            for (int c = 0; c < 4; ++c)
                *(v8bf*)&Ws[nxt][t][c * 8] = wreg[c];
#endif
        }
        wait_async0();
        __syncthreads();
    }

    // ---- epilogue: D layout M = v + 8*ks, N = m0; add bias, store f32 ----
    float bv[4];
#pragma unroll
    for (int j = 0; j < 4; ++j) bv[j] = bias[bN + wn + j * 16 + m0];

#pragma unroll
    for (int i = 0; i < 4; ++i) {
        const int row0 = bM + wm + i * 16 + ks * 8;
#pragma unroll
        for (int j = 0; j < 4; ++j) {
            const int col = bN + wn + j * 16 + m0;
            float* yp = Y + (size_t)row0 * NDIM + col;
#pragma unroll
            for (int v = 0; v < 8; ++v)
                yp[(size_t)v * NDIM] = acc[i][j][v] + bv[j];
        }
    }
}

// ---------------------------------------------------------------------------
// Host launcher. Inputs: x, core0, core1, core2, bias (all f32).
// Workspace: 4096*4096*2 = 32 MB for dense bf16 W.
// ---------------------------------------------------------------------------
extern "C" void kernel_launch(void* const* d_in, const int* in_sizes, int n_in,
                              void* d_out, int out_size, void* d_ws, size_t ws_size,
                              hipStream_t stream) {
    (void)in_sizes; (void)n_in; (void)out_size; (void)ws_size;
    const float* x    = (const float*)d_in[0];
    const float* c0   = (const float*)d_in[1];
    const float* c1   = (const float*)d_in[2];
    const float* c2   = (const float*)d_in[3];
    const float* bias = (const float*)d_in[4];
    float* y = (float*)d_out;
    __bf16* W = (__bf16*)d_ws;   // 32 MB

    tt_expand_w<<<(NDIM * (KDIM / 256)), 256, 0, stream>>>(c0, c1, c2, W);
    tt_gemm_bf16<<<dim3(NDIM / BN, MDIM / BM), 256, 0, stream>>>(x, W, bias, y);
}
